// GATEncoder_75720273428584
// MI455X (gfx1250) — matrix operand order repaired and checked by reading, not compiled, not run on hardware
//
#include <hip/hip_runtime.h>
#include <math.h>

// Problem constants (from reference)
constexpr int kN   = 50000;     // nodes
constexpr int kE   = 800000;    // real edges
constexpr int kDIN = 128;
constexpr int kED  = 16;
constexpr int kH   = 4;
constexpr int kC   = 128;
constexpr int kHC  = 512;       // H*C, GEMM N-dim (32 tiles of 16)
constexpr float kSlope = 0.2f;

constexpr int kMtiles    = kN / 16;              // 3125 (odd)
constexpr int kMtilesPad = ((kMtiles + 1) / 2) * 2;  // 3126, for 2x2 wave blocking
constexpr int kNtiles    = kHC / 16;             // 32

typedef __attribute__((ext_vector_type(16))) __bf16 v16bf;
typedef __attribute__((ext_vector_type(8)))  float  v8f;

// ---------------- helpers ----------------
__device__ __forceinline__ unsigned short f2bf(float f) {
  unsigned u = __float_as_uint(f);
  unsigned r = u + 0x7FFFu + ((u >> 16) & 1u);   // round-to-nearest-even
  return (unsigned short)(r >> 16);
}
__device__ __forceinline__ float lrelu(float a) { return a > 0.f ? a : kSlope * a; }
// order-preserving float<->uint for atomicMax
__device__ __forceinline__ unsigned encf(float f) {
  unsigned u = __float_as_uint(f);
  return (u & 0x80000000u) ? ~u : (u | 0x80000000u);
}
__device__ __forceinline__ float decf(unsigned e) {
  unsigned u = (e & 0x80000000u) ? (e & 0x7FFFFFFFu) : ~e;
  return __uint_as_float(u);
}
// WMMA bf16 fragment K-maps (wave32, 16x16x32)
__device__ __forceinline__ int kmapA(int lane, int j) {   // A 16x32: interleaved halves
  int kh = (lane & 16) ? 8 : 0;
  return (j < 8) ? (kh + j) : (8 + kh + j);
}
__device__ __forceinline__ int kmapB(int lane, int j) {   // B 32x16: interleaved halves (same as A)
  return kmapA(lane, j);
}

// ---------------- fills ----------------
__global__ void fill_f32(float* p, float v, long long n) {
  long long i = (long long)blockIdx.x * blockDim.x + threadIdx.x;
  if (i < n) p[i] = v;
}
__global__ void fill_u32(unsigned* p, unsigned v, long long n) {
  long long i = (long long)blockIdx.x * blockDim.x + threadIdx.x;
  if (i < n) p[i] = v;
}

// ---------------- tiny precomputes ----------------
__global__ void ea_sum_kernel(const float* __restrict__ ea, float* __restrict__ easum) {
  __shared__ float sh[kED];
  if (threadIdx.x < kED) sh[threadIdx.x] = 0.f;
  __syncthreads();
  int col = threadIdx.x & 15;
  float acc = 0.f;
  long long rpb = blockDim.x >> 4;
  for (long long r = (long long)blockIdx.x * rpb + (threadIdx.x >> 4); r < kE;
       r += (long long)gridDim.x * rpb)
    acc += ea[r * kED + col];
  atomicAdd(&sh[col], acc);
  __syncthreads();
  if (threadIdx.x < kED) atomicAdd(&easum[threadIdx.x], sh[threadIdx.x]);
}
// Me[d][h] = sum_c lin_e[d, h*C+c] * att_e[h, c]   (folds e=ea@We with a_e dot)
__global__ void make_me_kernel(const float* __restrict__ lin_e,
                               const float* __restrict__ att_e, float* __restrict__ Me) {
  int t = threadIdx.x;
  if (t >= kED * kH) return;
  int d = t >> 2, hh = t & 3;
  float s = 0.f;
  for (int c = 0; c < kC; ++c) s += lin_e[d * kHC + hh * kC + c] * att_e[hh * kC + c];
  Me[d * kH + hh] = s;
}
// al_loop[h] = mean_ea @ Me  (attention logit of every self-loop edge)
__global__ void make_alloop_kernel(const float* __restrict__ easum,
                                   const float* __restrict__ Me, float* __restrict__ al_loop) {
  int hh = threadIdx.x;
  if (hh >= kH) return;
  float s = 0.f;
  for (int d = 0; d < kED; ++d) s += (easum[d] * (1.0f / kE)) * Me[d * kH + hh];
  al_loop[hh] = s;
}

// ---------------- WMMA operand packing ----------------
// Wpack[nt][kt][lane][j] = bf16(W[k][n]), B fragment layout
__global__ void pack_W_kernel(const float* __restrict__ W, unsigned short* __restrict__ Wp, int K) {
  int Kt = K >> 5;
  int total = kNtiles * Kt * 32 * 16;
  for (int idx = blockIdx.x * blockDim.x + threadIdx.x; idx < total;
       idx += gridDim.x * blockDim.x) {
    int j = idx & 15, lane = (idx >> 4) & 31;
    int t = idx >> 9;
    int kt = t % Kt, nt = t / Kt;
    int n = nt * 16 + (lane & 15);
    int k = kt * 32 + kmapB(lane, j);
    Wp[idx] = f2bf(W[(long long)k * kHC + n]);
  }
}
// Apack[mt][kt][lane][j] = bf16(X[m][k]), A fragment layout; zero-fill pad tile
__global__ void pack_A_kernel(const float* __restrict__ X, unsigned short* __restrict__ Ap, int K) {
  int Kt = K >> 5;
  int total = kMtilesPad * Kt * 512;
  for (int idx = blockIdx.x * blockDim.x + threadIdx.x; idx < total;
       idx += gridDim.x * blockDim.x) {
    int j = idx & 15, lane = (idx >> 4) & 31;
    int t = idx >> 9;
    int kt = t % Kt, mt = t / Kt;
    int m = mt * 16 + (lane & 15);
    int k = kt * 32 + kmapA(lane, j);
    Ap[idx] = (m < kN) ? f2bf(X[(long long)m * K + k]) : (unsigned short)0;
  }
}

// ---------------- WMMA GEMM: h = X @ W, 2x2 16x16 tiles per wave ----------------
// 4 fragment loads feed 4 WMMAs per K-step (2x operand reuse vs 1x1 tiling).
__global__ void __launch_bounds__(256)
wmma_gemm_kernel(const unsigned short* __restrict__ Ap, const unsigned short* __restrict__ Wp,
                 float* __restrict__ Hout, int Kt) {
  int wave = threadIdx.x >> 5;
  int lane = threadIdx.x & 31;
  int job = blockIdx.x * 8 + wave;
  constexpr int kJobs = (kMtilesPad / 2) * (kNtiles / 2);   // 1563 * 16 = 25008
  if (job >= kJobs) return;
  int mt = (job >> 4) * 2;          // /(Ntiles/2)
  int nt = (job & 15) * 2;
  const unsigned short* a0 = Ap + ((long long)mt * Kt) * 512 + lane * 16;
  const unsigned short* a1 = a0 + (long long)Kt * 512;
  const unsigned short* b0 = Wp + ((long long)nt * Kt) * 512 + lane * 16;
  const unsigned short* b1 = b0 + (long long)Kt * 512;
  v8f c00 = {}, c01 = {}, c10 = {}, c11 = {};
  for (int kt = 0; kt < Kt; ++kt) {
    v16bf A0 = *(const v16bf*)(a0 + kt * 512);
    v16bf A1 = *(const v16bf*)(a1 + kt * 512);
    v16bf B0 = *(const v16bf*)(b0 + kt * 512);
    v16bf B1 = *(const v16bf*)(b1 + kt * 512);
    c00 = __builtin_amdgcn_wmma_f32_16x16x32_bf16(false, A0, false, B0, (short)0, c00, false, false);
    c01 = __builtin_amdgcn_wmma_f32_16x16x32_bf16(false, A0, false, B1, (short)0, c01, false, false);
    c10 = __builtin_amdgcn_wmma_f32_16x16x32_bf16(false, A1, false, B0, (short)0, c10, false, false);
    c11 = __builtin_amdgcn_wmma_f32_16x16x32_bf16(false, A1, false, B1, (short)0, c11, false, false);
  }
  int r0 = mt * 16 + ((lane >> 4) << 3);   // rows r0+v, v=0..7
  int n0 = nt * 16 + (lane & 15);
#pragma unroll
  for (int v = 0; v < 8; ++v) {
    int r = r0 + v;
    if (r < kN) {
      Hout[(long long)r * kHC + n0]      = c00[v];
      Hout[(long long)r * kHC + n0 + 16] = c01[v];
    }
    int r2 = r + 16;
    if (r2 < kN) {
      Hout[(long long)r2 * kHC + n0]      = c10[v];
      Hout[(long long)r2 * kHC + n0 + 16] = c11[v];
    }
  }
}

// ---------------- node attention logits ----------------
__global__ void node_al_kernel(const float* __restrict__ Hm, const float* __restrict__ att_src,
                               const float* __restrict__ att_dst, float* __restrict__ al_src,
                               float* __restrict__ al_dst) {
  int idx = blockIdx.x * blockDim.x + threadIdx.x;
  if (idx >= kN * kH) return;
  int n = idx >> 2, hh = idx & 3;
  const float* hp = Hm + (long long)n * kHC + hh * kC;
  const float* as = att_src + hh * kC;
  const float* ad = att_dst + hh * kC;
  float s = 0.f, d = 0.f;
#pragma unroll 4
  for (int c = 0; c < kC; ++c) { float hv = hp[c]; s += hv * as[c]; d += hv * ad[c]; }
  al_src[idx] = s;
  al_dst[idx] = d;
}

// ---------------- per-edge attention logits from edge features ----------------
__global__ void edge_ale_kernel(const float* __restrict__ ea, const float* __restrict__ Me,
                                float* __restrict__ al_e) {
  int e = blockIdx.x * blockDim.x + threadIdx.x;
  if (e >= kE) return;
  float v[kED];
#pragma unroll
  for (int d = 0; d < kED; ++d) v[d] = ea[(long long)e * kED + d];
#pragma unroll
  for (int hh = 0; hh < kH; ++hh) {
    float s = 0.f;
#pragma unroll
    for (int d = 0; d < kED; ++d) s += v[d] * Me[d * kH + hh];
    al_e[(long long)e * kH + hh] = s;
  }
}

// ---------------- segment softmax: pass 1 (raw logit + segment max) ----------------
__global__ void attn_max_kernel(const long long* __restrict__ ei, const float* __restrict__ al_src,
                                const float* __restrict__ al_dst, const float* __restrict__ al_e,
                                const float* __restrict__ al_loop, float* __restrict__ alphaE,
                                float* __restrict__ alphaS, unsigned* __restrict__ amax) {
  int i = blockIdx.x * blockDim.x + threadIdx.x;
  if (i >= kE + kN) return;
  if (i < kE) {
    int s = (int)ei[i], d2 = (int)ei[kE + i];
#pragma unroll
    for (int hh = 0; hh < kH; ++hh) {
      float a = lrelu(al_src[s * kH + hh] + al_dst[d2 * kH + hh] + al_e[(long long)i * kH + hh]);
      alphaE[(long long)i * kH + hh] = a;
      atomicMax(&amax[d2 * kH + hh], encf(a));
    }
  } else {
    int n = i - kE;
#pragma unroll
    for (int hh = 0; hh < kH; ++hh) {
      float a = lrelu(al_src[n * kH + hh] + al_dst[n * kH + hh] + al_loop[hh]);
      alphaS[n * kH + hh] = a;
      atomicMax(&amax[n * kH + hh], encf(a));
    }
  }
}

// ---------------- segment softmax: pass 2 (exp + denominator) ----------------
__global__ void attn_exp_kernel(const long long* __restrict__ ei, const unsigned* __restrict__ amax,
                                float* __restrict__ alphaE, float* __restrict__ alphaS,
                                float* __restrict__ den) {
  int i = blockIdx.x * blockDim.x + threadIdx.x;
  if (i >= kE + kN) return;
  int d2;
  float* ap;
  if (i < kE) { d2 = (int)ei[kE + i]; ap = alphaE + (long long)i * kH; }
  else        { d2 = i - kE;          ap = alphaS + (long long)d2 * kH; }
#pragma unroll
  for (int hh = 0; hh < kH; ++hh) {
    float ex = expf(ap[hh] - decf(amax[d2 * kH + hh]));
    ap[hh] = ex;
    atomicAdd(&den[d2 * kH + hh], ex);
  }
}

// ---------------- message aggregation: one wave per edge, coalesced atomics ----------------
__global__ void __launch_bounds__(256)
aggregate_kernel(const long long* __restrict__ ei, const float* __restrict__ alphaE,
                 const float* __restrict__ alphaS, const float* __restrict__ den,
                 const float* __restrict__ Hm, float* __restrict__ acc) {
  int wave = (blockIdx.x * blockDim.x + threadIdx.x) >> 5;
  int lane = threadIdx.x & 31;
  if (wave >= kE + kN) return;
  int s, d2;
  const float* ap;
  if (wave < kE) { s = (int)ei[wave]; d2 = (int)ei[kE + wave]; ap = alphaE + (long long)wave * kH; }
  else           { s = d2 = wave - kE;                         ap = alphaS + (long long)s * kH; }
  float sc[kH];
#pragma unroll
  for (int hh = 0; hh < kH; ++hh) sc[hh] = ap[hh] / (den[d2 * kH + hh] + 1e-16f);
  const float* hs = Hm + (long long)s * kHC;
  float* od = acc + (long long)d2 * kHC;
#pragma unroll
  for (int r = 0; r < 16; ++r) {
    int col = lane + 32 * r;            // coalesced across the wave
    atomicAdd(&od[col], sc[col >> 7] * hs[col]);
  }
}

// ---------------- bias + ELU epilogue ----------------
__global__ void elu_bias_kernel(const float* __restrict__ acc, const float* __restrict__ b,
                                float* __restrict__ out) {
  long long i = (long long)blockIdx.x * blockDim.x + threadIdx.x;
  if (i >= (long long)kN * kHC) return;
  float y = acc[i] + b[i & (kHC - 1)];
  out[i] = y > 0.f ? y : expm1f(y);
}

// ---------------- launch ----------------
extern "C" void kernel_launch(void* const* d_in, const int* in_sizes, int n_in,
                              void* d_out, int out_size, void* d_ws, size_t ws_size,
                              hipStream_t stream) {
  (void)in_sizes; (void)n_in; (void)out_size; (void)ws_size;
  const float*     x   = (const float*)d_in[0];
  const long long* ei  = (const long long*)d_in[1];
  const float*     ea  = (const float*)d_in[2];
  const float*     W1  = (const float*)d_in[3];
  const float*     as1 = (const float*)d_in[4];
  const float*     ad1 = (const float*)d_in[5];
  const float*     le1 = (const float*)d_in[6];
  const float*     ae1 = (const float*)d_in[7];
  const float*     b1  = (const float*)d_in[8];
  const float*     W2  = (const float*)d_in[9];
  const float*     as2 = (const float*)d_in[10];
  const float*     ad2 = (const float*)d_in[11];
  const float*     le2 = (const float*)d_in[12];
  const float*     ae2 = (const float*)d_in[13];
  const float*     b2  = (const float*)d_in[14];
  float* outf = (float*)d_out;

  size_t off = 0;
  auto take = [&](size_t bytes) -> char* {
    char* p = (char*)d_ws + off;
    off = (off + bytes + 255) & ~size_t(255);
    return p;
  };
  float*          h      = (float*)take((size_t)kN * kHC * 4);
  float*          out1   = (float*)take((size_t)kN * kHC * 4);
  unsigned short* Apack  = (unsigned short*)take((size_t)kMtilesPad * 16 * 512 * 2); // max Kt=16
  unsigned short* Wpack  = (unsigned short*)take((size_t)kHC * kHC * 2);
  float*          al_src = (float*)take((size_t)kN * kH * 4);
  float*          al_dst = (float*)take((size_t)kN * kH * 4);
  float*          al_e   = (float*)take((size_t)kE * kH * 4);
  float*          alphaE = (float*)take((size_t)kE * kH * 4);
  float*          alphaS = (float*)take((size_t)kN * kH * 4);
  unsigned*       amax   = (unsigned*)take((size_t)kN * kH * 4);
  float*          den    = (float*)take((size_t)kN * kH * 4);
  float*          easum  = (float*)take(64);
  float*          Me     = (float*)take(256);
  float*          alloop = (float*)take(64);

  fill_f32<<<1, 64, 0, stream>>>(easum, 0.f, kED);
  ea_sum_kernel<<<512, 256, 0, stream>>>(ea, easum);

  auto run_layer = [&](const float* X, int K, const float* W, const float* as, const float* ad,
                       const float* le, const float* aeh, const float* bias, float* layer_out) {
    int Kt = K >> 5;
    make_me_kernel<<<1, 64, 0, stream>>>(le, aeh, Me);
    make_alloop_kernel<<<1, 4, 0, stream>>>(easum, Me, alloop);
    pack_W_kernel<<<512, 256, 0, stream>>>(W, Wpack, K);
    pack_A_kernel<<<4096, 256, 0, stream>>>(X, Apack, K);
    constexpr int kJobs = (kMtilesPad / 2) * (kNtiles / 2);   // 25008
    wmma_gemm_kernel<<<(kJobs + 7) / 8, 256, 0, stream>>>(Apack, Wpack, h, Kt);
    node_al_kernel<<<(kN * kH + 255) / 256, 256, 0, stream>>>(h, as, ad, al_src, al_dst);
    edge_ale_kernel<<<(kE + 255) / 256, 256, 0, stream>>>(ea, Me, al_e);
    fill_u32<<<(kN * kH + 255) / 256, 256, 0, stream>>>(amax, 0u, (long long)kN * kH);
    fill_f32<<<(kN * kH + 255) / 256, 256, 0, stream>>>(den, 0.f, (long long)kN * kH);
    attn_max_kernel<<<(kE + kN + 255) / 256, 256, 0, stream>>>(ei, al_src, al_dst, al_e, alloop,
                                                               alphaE, alphaS, amax);
    attn_exp_kernel<<<(kE + kN + 255) / 256, 256, 0, stream>>>(ei, amax, alphaE, alphaS, den);
    fill_f32<<<(int)(((long long)kN * kHC + 255) / 256), 256, 0, stream>>>(outf, 0.f,
                                                                           (long long)kN * kHC);
    aggregate_kernel<<<(int)(((long long)(kE + kN) * 32 + 255) / 256), 256, 0, stream>>>(
        ei, alphaE, alphaS, den, h, outf);
    elu_bias_kernel<<<(int)(((long long)kN * kHC + 255) / 256), 256, 0, stream>>>(outf, bias,
                                                                                  layer_out);
  };

  run_layer(x,    kDIN, W1, as1, ad1, le1, ae1, b1, out1);   // layer 1 -> out1
  run_layer(out1, kHC,  W2, as2, ad2, le2, ae2, b2, outf);   // layer 2 -> d_out (in place)
}
